// TemporalMILBaginstanceLevel_85263690760703
// MI455X (gfx1250) — matrix operand order repaired and checked by reading, not compiled, not run on hardware
//
#include <hip/hip_runtime.h>

typedef __attribute__((ext_vector_type(16))) _Float16 v16h;
typedef __attribute__((ext_vector_type(8)))  _Float16 v8h;
typedef __attribute__((ext_vector_type(8)))  float    v8f;
typedef __attribute__((ext_vector_type(4)))  float    v4f;

#define B_   16
#define S_   128
#define H_   256
#define HID_ 128
#define G_   8      // blocks per batch for the window kernel
#define LDA  264    // padded row stride in halves (256 + 8): rows shift banks by 4 dwords

// ---- dynamic LDS layout (bytes) for the window kernel ----
#define OFF_W1T   0
#define SZ_W1T    (HID_ * LDA * 2)        // W1^T: 128 rows (n) x 264 halves
#define OFF_ATILE (OFF_W1T + SZ_W1T)
#define SZ_ATILE  (8 * 16 * LDA * 2)      // 8 waves x 16 rows x 264 halves
#define OFF_B1    (OFF_ATILE + SZ_ATILE)
#define SZ_B1     (HID_ * 4)
#define OFF_W2    (OFF_B1 + SZ_B1)
#define SZ_W2     (HID_ * 4)
#define OFF_RM    (OFF_W2 + SZ_W2)
#define SZ_RM     (8 * 4)
#define OFF_RD    (OFF_RM + SZ_RM)
#define SZ_RD     (8 * 4)
#define OFF_RN    (OFF_RD + SZ_RD)
#define SZ_RN     (8 * H_ * 4)
#define LDS_TOTAL (OFF_RN + SZ_RN)        // = 144448 B  (< 320 KB WGP LDS)

// =====================================================================
// Kernel 1: instance-level attention pooling (tiny: ~68 MMAC, scalar VALU)
// =====================================================================
__global__ __launch_bounds__(128, 1)
void mil_instance_kernel(const float* __restrict__ x,
                         const float* __restrict__ wi1,
                         const float* __restrict__ bi1,
                         const float* __restrict__ wi2,
                         const float* __restrict__ bi2,
                         float* __restrict__ instRep)
{
  __shared__ float sc[S_];
  __shared__ float red[4];
  const int b    = blockIdx.x;
  const int j    = threadIdx.x;       // 0..127 -> hidden unit
  const int wave = j >> 5, lane = j & 31;
  const float w2v = wi2[j];
  const float b1v = bi1[j];
  const float b2v = bi2[0];

  for (int s = 0; s < S_; ++s) {
    const float* xr = x + ((size_t)(b * S_ + s)) * H_;
    float h = b1v;
    for (int k = 0; k < H_; ++k) h += xr[k] * wi1[k * HID_ + j];
    float v = fmaxf(h, 0.f) * w2v;
    v += __shfl_xor(v, 1, 32);
    v += __shfl_xor(v, 2, 32);
    v += __shfl_xor(v, 4, 32);
    v += __shfl_xor(v, 8, 32);
    v += __shfl_xor(v, 16, 32);
    if (lane == 0) red[wave] = v;
    __syncthreads();
    if (j == 0) sc[s] = red[0] + red[1] + red[2] + red[3] + b2v;
    __syncthreads();
  }

  // softmax over sc[0..127]
  float mx = -__builtin_inff();
  for (int s = 0; s < S_; ++s) mx = fmaxf(mx, sc[s]);
  float sum = 0.f;
  for (int s = 0; s < S_; ++s) sum += __expf(sc[s] - mx);
  float pj = __expf(sc[j] - mx) / sum;
  __syncthreads();
  sc[j] = pj;
  __syncthreads();

  for (int hh = j; hh < H_; hh += 128) {
    float acc = 0.f;
    for (int s = 0; s < S_; ++s)
      acc += sc[s] * x[((size_t)(b * S_ + s)) * H_ + hh];
    instRep[b * H_ + hh] = acc;
  }
}

// =====================================================================
// Kernel 2: fused window max-scan + MLP (WMMA f16) + online softmax.
// Grid: 16 batches x 8 blocks; block = 256 threads = 8 waves.
// Wave-unit u handles chains s=u and s=126-u (constant 128 windows/unit).
// =====================================================================
__global__ __launch_bounds__(256, 1)
void mil_window_kernel(const float* __restrict__ x,
                       const float* __restrict__ ww1,
                       const float* __restrict__ bw1,
                       const float* __restrict__ ww2,
                       const float* __restrict__ bw2,
                       float* __restrict__ part)   // [B][G][2 + H]
{
  extern __shared__ char smem[];
  _Float16* w1t  = (_Float16*)(smem + OFF_W1T);
  _Float16* atil = (_Float16*)(smem + OFF_ATILE);
  float* b1s  = (float*)(smem + OFF_B1);
  float* w2s  = (float*)(smem + OFF_W2);
  float* redM = (float*)(smem + OFF_RM);
  float* redD = (float*)(smem + OFF_RD);
  float* redN = (float*)(smem + OFF_RN);

  const int tid  = threadIdx.x;
  const int wave = tid >> 5;
  const int lane = tid & 31;
  const int b = blockIdx.x >> 3;
  const int g = blockIdx.x & 7;

  // Stage W1^T (f16, n-major so K is contiguous per B-fragment), b1, w2.
  for (int idx = tid; idx < H_ * HID_; idx += 256) {
    int h = idx >> 7;         // 0..255
    int n = idx & 127;        // 0..127
    w1t[n * LDA + h] = (_Float16)ww1[idx];
  }
  for (int n = tid; n < HID_; n += 256) { b1s[n] = bw1[n]; w2s[n] = ww2[n]; }
  __syncthreads();

  const float b2v = bw2[0];
  _Float16* aw = atil + wave * 16 * LDA;  // this wave's 16x256 f16 A-tile

  // Hoist per-lane bias / w2 values (depend only on nt and lane&15).
  float b1r[8], w2r[8];
  #pragma unroll
  for (int nt = 0; nt < 8; ++nt) {
    b1r[nt] = b1s[nt * 16 + (lane & 15)];
    w2r[nt] = w2s[nt * 16 + (lane & 15)];
  }

  float num[8];
  #pragma unroll
  for (int j = 0; j < 8; ++j) num[j] = 0.f;
  float denom = 0.f;
  float Mx = -__builtin_inff();

  const int unit = g * 8 + wave;       // 0..63
  const int sA = unit, sB = 126 - unit;

  for (int c = 0; c < 2; ++c) {
    if (c == 1 && sB == sA) break;
    const int s = (c == 0) ? sA : sB;

    // running max m = x[b, s, :]; lane owns h = lane*8 .. lane*8+7
    float m[8];
    {
      const v4f* p = (const v4f*)(x + ((size_t)(b * S_ + s)) * H_ + lane * 8);
      v4f a0 = p[0], a1 = p[1];
      #pragma unroll
      for (int j = 0; j < 4; ++j) { m[j] = a0[j]; m[4 + j] = a1[j]; }
    }

    for (int e0 = s + 1; e0 < S_; e0 += 16) {
      const int rows = (S_ - e0 < 16) ? (S_ - e0) : 16;

      asm volatile("" ::: "memory");
      // Build 16 pooled rows (f16) into the LDS A-tile (16B store per lane).
      #pragma unroll
      for (int r = 0; r < 16; ++r) {
        if (r < rows) {
          const v4f* p = (const v4f*)(x + ((size_t)(b * S_ + e0 + r)) * H_ + lane * 8);
          v4f a0 = p[0], a1 = p[1];
          #pragma unroll
          for (int j = 0; j < 4; ++j) {
            m[j]     = fmaxf(m[j],     a0[j]);
            m[4 + j] = fmaxf(m[4 + j], a1[j]);
          }
        }
        v8h hr;
        #pragma unroll
        for (int j = 0; j < 8; ++j) hr[j] = (_Float16)m[j];
        *(v8h*)(aw + r * LDA + lane * 8) = hr;
      }
      asm volatile("s_wait_dscnt 0" ::: "memory");  // wave-local LDS RAW order

      // GEMM: D[16x128] = A[16x256] x W1[256x128] (f32 accum, bias at epilogue).
      // kt = 0 uses a constant-zero C so the backend emits inline SRC2 = 0.
      v8f acc[8];
      const _Float16* arow = aw + (lane & 15) * LDA;
      {
        const int ka = ((lane >> 4) << 4);
        v8h alo = *(const v8h*)(arow + ka);
        v8h ahi = *(const v8h*)(arow + ka + 8);
        v16h af;
        #pragma unroll
        for (int i = 0; i < 8; ++i) { af[i] = alo[i]; af[8 + i] = ahi[i]; }
        const int kb0 = ((lane >> 4) << 4);
        v8f zero;
        #pragma unroll
        for (int i = 0; i < 8; ++i) zero[i] = 0.f;
        #pragma unroll
        for (int nt = 0; nt < 8; ++nt) {
          const _Float16* wrow = w1t + (nt * 16 + (lane & 15)) * LDA + kb0;
          v8h blo = *(const v8h*)(wrow);
          v8h bhi = *(const v8h*)(wrow + 8);
          v16h bf;
          #pragma unroll
          for (int i = 0; i < 8; ++i) { bf[i] = blo[i]; bf[8 + i] = bhi[i]; }
          acc[nt] = __builtin_amdgcn_wmma_f32_16x16x32_f16(
                        false, af, false, bf, (short)0, zero, false, false);
        }
      }
      #pragma unroll
      for (int kt = 1; kt < 8; ++kt) {
        // A frag: M = lane&15, K contiguous; lanes>=16 K += 16
        const int ka = kt * 32 + ((lane >> 4) << 4);
        v8h alo = *(const v8h*)(arow + ka);
        v8h ahi = *(const v8h*)(arow + ka + 8);
        v16h af;
        #pragma unroll
        for (int i = 0; i < 8; ++i) { af[i] = alo[i]; af[8 + i] = ahi[i]; }
        // B frag: N = lane&15, K contiguous; lanes>=16 K += 16
        const int kb0 = kt * 32 + ((lane >> 4) << 4);
        #pragma unroll
        for (int nt = 0; nt < 8; ++nt) {
          const _Float16* wrow = w1t + (nt * 16 + (lane & 15)) * LDA + kb0;
          v8h blo = *(const v8h*)(wrow);
          v8h bhi = *(const v8h*)(wrow + 8);
          v16h bf;
          #pragma unroll
          for (int i = 0; i < 8; ++i) { bf[i] = blo[i]; bf[8 + i] = bhi[i]; }
          acc[nt] = __builtin_amdgcn_wmma_f32_16x16x32_f16(
                        false, af, false, bf, (short)0, acc[nt], false, false);
        }
      }

      // score_r = sum_n relu(D[r][n] + b1[n]) * w2[n] + b2 ; C/D layout: lane
      // holds N = nt*16 + (lane&15), row M = r + 8*(lane>=16).
      float prt[8];
      #pragma unroll
      for (int r = 0; r < 8; ++r) prt[r] = 0.f;
      #pragma unroll
      for (int nt = 0; nt < 8; ++nt) {
        const float w2v = w2r[nt];
        const float b1v = b1r[nt];
        #pragma unroll
        for (int r = 0; r < 8; ++r) prt[r] += fmaxf(acc[nt][r] + b1v, 0.f) * w2v;
      }
      float sc[16];
      #pragma unroll
      for (int r = 0; r < 8; ++r) {
        float v = prt[r];
        v += __shfl_xor(v, 1, 32);
        v += __shfl_xor(v, 2, 32);
        v += __shfl_xor(v, 4, 32);
        v += __shfl_xor(v, 8, 32);
        v += b2v;
        const float o = __shfl_xor(v, 16, 32);
        if (lane < 16) { sc[r] = v; sc[r + 8] = o; }
        else           { sc[r] = o; sc[r + 8] = v; }
      }

      // Online softmax accumulate (valid rows only).
      float tmax = -__builtin_inff();
      for (int r = 0; r < rows; ++r) tmax = fmaxf(tmax, sc[r]);
      const float newM = fmaxf(Mx, tmax);
      const float resc = __expf(Mx - newM);
      denom *= resc;
      #pragma unroll
      for (int j = 0; j < 8; ++j) num[j] *= resc;
      for (int r = 0; r < rows; ++r) {
        const float w = __expf(sc[r] - newM);
        denom += w;
        v8h mr = *(const v8h*)(aw + r * LDA + lane * 8);
        #pragma unroll
        for (int j = 0; j < 8; ++j) num[j] += w * (float)mr[j];
      }
      Mx = newM;
    }
  }

  // Block-level combine of the 8 wave partials.
  if (lane == 0) { redM[wave] = Mx; redD[wave] = denom; }
  #pragma unroll
  for (int j = 0; j < 8; ++j) redN[wave * H_ + lane * 8 + j] = num[j];
  __syncthreads();

  float bM = -__builtin_inff();
  #pragma unroll
  for (int w = 0; w < 8; ++w) bM = fmaxf(bM, redM[w]);
  float ns = 0.f;
  #pragma unroll
  for (int w = 0; w < 8; ++w) ns += redN[w * H_ + tid] * __expf(redM[w] - bM);
  float* pb = part + ((size_t)(b * G_ + g)) * (2 + H_);
  pb[2 + tid] = ns;
  if (tid == 0) {
    float ds = 0.f;
    #pragma unroll
    for (int w = 0; w < 8; ++w) ds += redD[w] * __expf(redM[w] - bM);
    pb[0] = bM; pb[1] = ds;
  }
}

// =====================================================================
// Kernel 3: merge block partials per batch, add instance path, /2.
// =====================================================================
__global__ __launch_bounds__(256, 1)
void mil_combine_kernel(const float* __restrict__ instRep,
                        const float* __restrict__ part,
                        float* __restrict__ out)
{
  const int b = blockIdx.x;
  const int h = threadIdx.x;
  float gM = -__builtin_inff();
  for (int g = 0; g < G_; ++g)
    gM = fmaxf(gM, part[((size_t)(b * G_ + g)) * (2 + H_)]);
  float ds = 0.f, ns = 0.f;
  for (int g = 0; g < G_; ++g) {
    const float* pb = part + ((size_t)(b * G_ + g)) * (2 + H_);
    const float scl = __expf(pb[0] - gM);
    ds += pb[1] * scl;
    ns += pb[2 + h] * scl;
  }
  out[b * H_ + h] = 0.5f * (instRep[b * H_ + h] + ns / ds);
}

// =====================================================================
extern "C" void kernel_launch(void* const* d_in, const int* in_sizes, int n_in,
                              void* d_out, int out_size, void* d_ws, size_t ws_size,
                              hipStream_t stream) {
  (void)in_sizes; (void)n_in; (void)out_size; (void)ws_size;
  const float* x   = (const float*)d_in[0];
  const float* wi1 = (const float*)d_in[1];
  const float* bi1 = (const float*)d_in[2];
  const float* wi2 = (const float*)d_in[3];
  const float* bi2 = (const float*)d_in[4];
  const float* ww1 = (const float*)d_in[5];
  const float* bw1 = (const float*)d_in[6];
  const float* ww2 = (const float*)d_in[7];
  const float* bw2 = (const float*)d_in[8];

  float* instRep = (float*)d_ws;                 // [16][256]
  float* part    = instRep + B_ * H_;            // [16][8][2+256]

  mil_instance_kernel<<<B_, 128, 0, stream>>>(x, wi1, bi1, wi2, bi2, instRep);
  mil_window_kernel<<<B_ * G_, 256, LDS_TOTAL, stream>>>(x, ww1, bw1, ww2, bw2, part);
  mil_combine_kernel<<<B_, 256, 0, stream>>>(instRep, part, (float*)d_out);
}
